// FlashAttention_13254269075497
// MI455X (gfx1250) — matrix operand order repaired and checked by reading, not compiled, not run on hardware
//
#include <hip/hip_runtime.h>

// ---------------- types ----------------
typedef __attribute__((ext_vector_type(16))) __bf16        v16bf;
typedef __attribute__((ext_vector_type(8)))  float         v8f;
typedef __attribute__((ext_vector_type(4)))  float         f32x4;
typedef __attribute__((ext_vector_type(4)))  unsigned int  u32x4;
typedef __attribute__((ext_vector_type(2)))  unsigned int  u32x2;

union Frag16 {
    v16bf          bf;
    unsigned short us[16];
    u32x4          q[2];
};

#define SEQ_LEN   8192
#define HEAD_DIM  128
#define BM        64           // query rows per block (16 per wave)
#define BN        64           // K/V rows per iteration
#define NW        4            // waves per block
#define NITER     (SEQ_LEN / BN)

__device__ __forceinline__ unsigned short f2bf(float f) {
    union { float f; unsigned int u; } c; c.f = f;
    unsigned int u = c.u;
    unsigned int r = u + 0x7FFFu + ((u >> 16) & 1u);   // round-to-nearest-even
    return (unsigned short)(r >> 16);
}

// Async DMA: global (bf16, 16B chunk) -> LDS. ASYNCcnt-tracked.
__device__ __forceinline__ void async_cp16(unsigned lds_addr, const unsigned short* g) {
    unsigned long long ga = (unsigned long long)g;
    asm volatile("global_load_async_to_lds_b128 %0, %1, off"
                 :: "v"(lds_addr), "v"(ga) : "memory");
}

// ============ prep: K -> bf16 row-major, V -> bf16 transposed ============
__global__ void fa_prep_kernel(const float* __restrict__ Kg,
                               const float* __restrict__ Vg,
                               unsigned short* __restrict__ Kbf,   // [SEQ][128]
                               unsigned short* __restrict__ Vtg) { // [128][SEQ]
    const int i  = blockIdx.x * blockDim.x + threadIdx.x;  // over SEQ*128/4
    const int n  = i >> 5;                                 // seq row
    const int d4 = (i & 31) << 2;                          // head-dim *4
    const f32x4 kf = *(const f32x4*)(Kg + n * HEAD_DIM + d4);
    union { unsigned short us[4]; u32x2 q; } pk;
    pk.us[0] = f2bf(kf.x); pk.us[1] = f2bf(kf.y);
    pk.us[2] = f2bf(kf.z); pk.us[3] = f2bf(kf.w);
    *(u32x2*)(Kbf + n * HEAD_DIM + d4) = pk.q;

    const f32x4 vf = *(const f32x4*)(Vg + n * HEAD_DIM + d4);
    Vtg[(d4 + 0) * SEQ_LEN + n] = f2bf(vf.x);
    Vtg[(d4 + 1) * SEQ_LEN + n] = f2bf(vf.y);
    Vtg[(d4 + 2) * SEQ_LEN + n] = f2bf(vf.z);
    Vtg[(d4 + 3) * SEQ_LEN + n] = f2bf(vf.w);
}

// ============ main flash-attention kernel ============
__launch_bounds__(NW * 32, 1)
__global__ void fa_fwd_kernel(const float* __restrict__ Qg,
                              const unsigned short* __restrict__ Kbf,
                              const unsigned short* __restrict__ Vtg,
                              float* __restrict__ Og) {
    // Double-buffered bf16 tiles + per-wave P staging
    __shared__ __align__(16) unsigned short KsL[2][BN * HEAD_DIM]; // 2x16 KB [n][d]
    __shared__ __align__(16) unsigned short VtL[2][HEAD_DIM * BN]; // 2x16 KB [d][n]
    __shared__ __align__(16) unsigned short Ps[NW][BN * 16];       //   8 KB [k][m]

    const int tid  = threadIdx.x;
    const int lane = tid & 31;
    const int wave = tid >> 5;
    const int hl   = lane >> 4;
    const int l16  = lane & 15;

    const int qBase = blockIdx.x * BM + wave * 16;

    // ---- async prefetch of one K/V tile (16 x 16B chunks per thread) ----
    auto prefetch_tile = [&](int buf, int nBase) {
        #pragma unroll
        for (int c = 0; c < 8; ++c) {
            const int chunk = tid + c * 128;                 // 1024 chunks of 16B
            const unsigned short* gk = Kbf + nBase * HEAD_DIM + chunk * 8;
            unsigned lk = (unsigned)(unsigned long long)(const void*)&KsL[buf][chunk * 8];
            async_cp16(lk, gk);
        }
        #pragma unroll
        for (int c = 0; c < 8; ++c) {
            const int chunk = tid + c * 128;
            const int d = chunk >> 3, o = (chunk & 7) * 8;   // 8 chunks per d-row
            const unsigned short* gv = Vtg + d * SEQ_LEN + nBase + o;
            unsigned lv = (unsigned)(unsigned long long)(const void*)&VtL[buf][d * BN + o];
            async_cp16(lv, gv);
        }
    };

    // ---- load Q strip into A-fragment layout (bf16), kept in registers ----
    v16bf qf[4];
    {
        const int row = qBase + l16;
        #pragma unroll
        for (int c = 0; c < 4; ++c) {
            Frag16 f;
            const int kb = c * 32 + (hl ? 8 : 0);
            const float* src = Qg + row * HEAD_DIM + kb;
            #pragma unroll
            for (int i = 0; i < 8; ++i) f.us[i]     = f2bf(src[i]);
            #pragma unroll
            for (int i = 0; i < 8; ++i) f.us[8 + i] = f2bf(src[16 + i]);
            qf[c] = f.bf;
        }
    }

    // ---- running flash state (C/D layout: slot r -> row r + 8*hl) ----
    float mrow[8], lrow[8];
    v8f o[8];
    #pragma unroll
    for (int r = 0; r < 8; ++r) { mrow[r] = -3.0e38f; lrow[r] = 0.0f; }
    #pragma unroll
    for (int t = 0; t < 8; ++t) { v8f z = {}; o[t] = z; }

    prefetch_tile(0, 0);

    for (int it = 0; it < NITER; ++it) {
        const int cur = it & 1;

        if (it + 1 < NITER) {
            prefetch_tile(cur ^ 1, (it + 1) * BN);
            // allow the 16 just-issued ops; tile `it` (in-order) is complete
            asm volatile("s_wait_asynccnt 0x10" ::: "memory");
        } else {
            asm volatile("s_wait_asynccnt 0x0" ::: "memory");
        }
        __syncthreads();

        // ---- S = Q * K^T : 16 x 64 per wave (16 WMMAs) ----
        v8f s[4];
        #pragma unroll
        for (int j = 0; j < 4; ++j) {
            v8f acc = {};
            #pragma unroll
            for (int c = 0; c < 4; ++c) {
                Frag16 b;
                const unsigned short* src =
                    &KsL[cur][(j * 16 + l16) * HEAD_DIM + c * 32 + (hl ? 8 : 0)];
                b.q[0] = *(const u32x4*)(src);
                b.q[1] = *(const u32x4*)(src + 16);
                acc = __builtin_amdgcn_wmma_f32_16x16x32_bf16(
                          false, qf[c], false, b.bf, (short)0, acc, false, false);
            }
            s[j] = acc;
        }

        // ---- online softmax (row reductions across 16-lane halves) ----
        float scale[8], pr[4][8];
        #pragma unroll
        for (int r = 0; r < 8; ++r) {
            float mx = s[0][r];
            #pragma unroll
            for (int j = 1; j < 4; ++j) mx = fmaxf(mx, s[j][r]);
            #pragma unroll
            for (int off = 1; off < 16; off <<= 1)
                mx = fmaxf(mx, __shfl_xor(mx, off, 32));
            const float mnew = fmaxf(mrow[r], mx);
            scale[r] = __expf(mrow[r] - mnew);
            mrow[r]  = mnew;
            float sum = 0.0f;
            #pragma unroll
            for (int j = 0; j < 4; ++j) {
                const float p = __expf(s[j][r] - mnew);
                pr[j][r] = p;
                sum += p;
            }
            #pragma unroll
            for (int off = 1; off < 16; off <<= 1)
                sum += __shfl_xor(sum, off, 32);
            lrow[r] = lrow[r] * scale[r] + sum;
        }

        // ---- rescale O accumulators ----
        #pragma unroll
        for (int t = 0; t < 8; ++t) {
            #pragma unroll
            for (int r = 0; r < 8; ++r) o[t][r] *= scale[r];
        }

        // ---- stage P (bf16) to LDS in [k][m] layout ----
        #pragma unroll
        for (int j = 0; j < 4; ++j) {
            union { unsigned short us[8]; u32x4 q; } w;
            #pragma unroll
            for (int r = 0; r < 8; ++r) w.us[r] = f2bf(pr[j][r]);
            *(u32x4*)(&Ps[wave][(j * 16 + l16) * 16 + hl * 8]) = w.q;
        }
        __syncthreads();

        // ---- O += P * V : 16 WMMAs ----
        #pragma unroll
        for (int c = 0; c < 2; ++c) {
            Frag16 a;
            const int kb = c * 32 + (hl ? 8 : 0);
            #pragma unroll
            for (int i = 0; i < 8; ++i) a.us[i]     = Ps[wave][(kb + i) * 16 + l16];
            #pragma unroll
            for (int i = 0; i < 8; ++i) a.us[8 + i] = Ps[wave][(kb + 16 + i) * 16 + l16];
            #pragma unroll
            for (int t = 0; t < 8; ++t) {
                Frag16 b;
                const unsigned short* src =
                    &VtL[cur][(t * 16 + l16) * BN + c * 32 + (hl ? 8 : 0)];
                b.q[0] = *(const u32x4*)(src);
                b.q[1] = *(const u32x4*)(src + 16);
                o[t] = __builtin_amdgcn_wmma_f32_16x16x32_bf16(
                           false, a.bf, false, b.bf, (short)0, o[t], false, false);
            }
        }
        __syncthreads();   // all waves done with buffer `cur` before it is refilled
    }

    // ---- normalize and store O (f32) ----
    #pragma unroll
    for (int r = 0; r < 8; ++r) {
        const float inv = 1.0f / lrow[r];
        const int row = qBase + hl * 8 + r;
        #pragma unroll
        for (int t = 0; t < 8; ++t)
            Og[row * HEAD_DIM + t * 16 + l16] = o[t][r] * inv;
    }
}

extern "C" void kernel_launch(void* const* d_in, const int* in_sizes, int n_in,
                              void* d_out, int out_size, void* d_ws, size_t ws_size,
                              hipStream_t stream) {
    const float* q = (const float*)d_in[0];
    const float* k = (const float*)d_in[1];
    const float* v = (const float*)d_in[2];
    float* out = (float*)d_out;

    unsigned short* Kbf = (unsigned short*)d_ws;                 // 2 MB
    unsigned short* Vtg = Kbf + SEQ_LEN * HEAD_DIM;              // 2 MB

    dim3 pgrid((SEQ_LEN * HEAD_DIM / 4) / 256);
    fa_prep_kernel<<<pgrid, 256, 0, stream>>>(k, v, Kbf, Vtg);

    dim3 grid(SEQ_LEN / BM);
    dim3 block(NW * 32);
    fa_fwd_kernel<<<grid, block, 0, stream>>>(q, Kbf, Vtg, out);
}
